// PointNetSetAbstractionKNN_83425444757843
// MI455X (gfx1250) — matrix operand order repaired and checked by reading, not compiled, not run on hardware
//
#include <hip/hip_runtime.h>

#define NPTS  4096
#define BATCH 16
#define SPTS  1024
#define KNN   32
#define CIN   64
#define MROWS (BATCH*SPTS*KNN)   // 524288

typedef __attribute__((ext_vector_type(16))) _Float16 v16h;
typedef __attribute__((ext_vector_type(8)))  _Float16 v8h;
typedef __attribute__((ext_vector_type(8)))  float    v8f;

// ---------------------------------------------------------------------------
// Weight prep: transpose to [C][O] layout (B-fragment wants N contiguous per
// fixed K=row), cast to f16, pad layer-0 input channels 67 -> 96.
// ---------------------------------------------------------------------------
__global__ void convert_weights(const float* __restrict__ W0, const float* __restrict__ W1,
                                const float* __restrict__ W2, _Float16* __restrict__ Wt0,
                                _Float16* __restrict__ Wt1, _Float16* __restrict__ Wt2) {
  int t = blockIdx.x * blockDim.x + threadIdx.x;
  if (t < 96 * 64) {
    int c = t >> 6, o = t & 63;
    Wt0[t] = (c < 67) ? (_Float16)W0[o * 67 + c] : (_Float16)0.f;
  } else if (t < 96 * 64 + 64 * 64) {
    int e = t - 96 * 64;
    int c = e >> 6, o = e & 63;
    Wt1[e] = (_Float16)W1[o * 64 + c];
  } else if (t < 96 * 64 + 64 * 64 + 64 * 128) {
    int e = t - (96 * 64 + 64 * 64);
    int c = e >> 7, o = e & 127;
    Wt2[e] = (_Float16)W2[o * 64 + c];
  }
}

__global__ void zero_f32(float* __restrict__ p, int n) {
  int t = blockIdx.x * blockDim.x + threadIdx.x;
  if (t < n) p[t] = 0.f;
}

// ---------------------------------------------------------------------------
// Farthest point sampling: one workgroup per batch; points + running min-dist
// live in LDS (64KB + reduction scratch, fine under CDNA5's 320KB/WGP).
// ---------------------------------------------------------------------------
__global__ __launch_bounds__(256) void fps_kernel(const float* __restrict__ xyz,
                                                  int* __restrict__ fps_idx,
                                                  float* __restrict__ new_xyz,
                                                  float* __restrict__ out_xyz_t) {
  __shared__ float s_px[NPTS], s_py[NPTS], s_pz[NPTS], s_dist[NPTS];
  __shared__ float s_red[256];
  __shared__ int   s_redi[256];
  int b = blockIdx.x, tid = threadIdx.x;
  const float* base = xyz + (size_t)b * 3 * NPTS;
  for (int n = tid; n < NPTS; n += 256) {
    s_px[n] = base[n];
    s_py[n] = base[NPTS + n];
    s_pz[n] = base[2 * NPTS + n];
    s_dist[n] = 1e10f;
  }
  __syncthreads();
  int far = 0;
  for (int s = 0; s < SPTS; ++s) {
    float cx = s_px[far], cy = s_py[far], cz = s_pz[far];
    if (tid == 0) {
      fps_idx[b * SPTS + s] = far;
      float* nx = new_xyz + (size_t)(b * SPTS + s) * 3;
      nx[0] = cx; nx[1] = cy; nx[2] = cz;
      out_xyz_t[(size_t)b * 3 * SPTS + 0 * SPTS + s] = cx;
      out_xyz_t[(size_t)b * 3 * SPTS + 1 * SPTS + s] = cy;
      out_xyz_t[(size_t)b * 3 * SPTS + 2 * SPTS + s] = cz;
    }
    float bd = -1.f; int bi = 0;
    for (int n = tid; n < NPTS; n += 256) {
      float dx = s_px[n] - cx, dy = s_py[n] - cy, dz = s_pz[n] - cz;
      float d = dx * dx + dy * dy + dz * dz;
      float m = fminf(s_dist[n], d);
      s_dist[n] = m;
      if (m > bd) { bd = m; bi = n; }
    }
    s_red[tid] = bd; s_redi[tid] = bi;
    __syncthreads();
    for (int off = 128; off > 0; off >>= 1) {
      if (tid < off && s_red[tid + off] > s_red[tid]) {
        s_red[tid] = s_red[tid + off];
        s_redi[tid] = s_redi[tid + off];
      }
      __syncthreads();
    }
    far = s_redi[0];
    __syncthreads();
  }
}

// ---------------------------------------------------------------------------
// KNN: one wave32 per query, one (dist,idx) candidate per lane, bitonic
// sort/merge over shfl_xor. Lane l ends with the l-th nearest neighbor.
// ---------------------------------------------------------------------------
__device__ __forceinline__ void bitonic_sort32(float& d, int& i) {
  int lane = threadIdx.x & 31;
  for (int kk = 2; kk <= 32; kk <<= 1) {
    for (int j = kk >> 1; j > 0; j >>= 1) {
      float od = __shfl_xor(d, j, 32);
      int   oi = __shfl_xor(i, j, 32);
      bool up = ((lane & kk) == 0);
      bool lower = ((lane & j) == 0);
      bool smaller = od < d;
      bool take = up ? (lower ? smaller : !smaller) : (lower ? !smaller : smaller);
      if (take) { d = od; i = oi; }
    }
  }
}

__device__ __forceinline__ void bitonic_merge32(float& d, int& i) {
  int lane = threadIdx.x & 31;
  for (int j = 16; j > 0; j >>= 1) {
    float od = __shfl_xor(d, j, 32);
    int   oi = __shfl_xor(i, j, 32);
    bool lower = ((lane & j) == 0);
    bool take = lower ? (od < d) : (od > d);
    if (take) { d = od; i = oi; }
  }
}

__global__ __launch_bounds__(256) void knn_kernel(const float* __restrict__ xyz,
                                                  const float* __restrict__ new_xyz,
                                                  int* __restrict__ knn_idx) {
  int w = (blockIdx.x * blockDim.x + threadIdx.x) >> 5;   // query id, grid sized exactly
  int lane = threadIdx.x & 31;
  int b = w / SPTS;
  const float* q = new_xyz + (size_t)w * 3;
  float qx = q[0], qy = q[1], qz = q[2];
  const float* base = xyz + (size_t)b * 3 * NPTS;

  int n = lane;
  float dx = base[n] - qx, dy = base[NPTS + n] - qy, dz = base[2 * NPTS + n] - qz;
  float bd = dx * dx + dy * dy + dz * dz;
  int bi = n;
  bitonic_sort32(bd, bi);

  for (int c = 1; c < NPTS / 32; ++c) {
    n = c * 32 + lane;
    dx = base[n] - qx; dy = base[NPTS + n] - qy; dz = base[2 * NPTS + n] - qz;
    float d = dx * dx + dy * dy + dz * dz;
    float kth = __shfl(bd, 31, 32);
    if (__any(d < kth)) {
      int ci = n;
      bitonic_sort32(d, ci);
      float rd = __shfl(d, 31 - lane, 32);   // reverse -> descending
      int   ri = __shfl(ci, 31 - lane, 32);
      if (rd < bd) { bd = rd; bi = ri; }     // keep 32 smallest (bitonic)
      bitonic_merge32(bd, bi);               // re-sort ascending
    }
  }
  knn_idx[(size_t)w * KNN + lane] = bi;
}

// ---------------------------------------------------------------------------
// Build f16 im2row matrix A0 [MROWS x 96]: cols 0..2 = grouped_xyz delta,
// 3..66 = gathered features, 67..95 = zero pad (so WMMA K-steps are 3 x 32).
// ---------------------------------------------------------------------------
__global__ void gather_kernel(const float* __restrict__ xyz, const float* __restrict__ points,
                              const float* __restrict__ new_xyz, const int* __restrict__ knn_idx,
                              _Float16* __restrict__ A0) {
  size_t t = (size_t)blockIdx.x * blockDim.x + threadIdx.x;
  if (t >= (size_t)MROWS * 96) return;
  int c = (int)(t % 96);
  size_t r = t / 96;
  int bs = (int)(r / KNN);     // b*S + s
  int b = bs / SPTS;
  int n = knn_idx[r];
  float v;
  if (c < 3)       v = xyz[(size_t)b * 3 * NPTS + (size_t)c * NPTS + n] - new_xyz[(size_t)bs * 3 + c];
  else if (c < 67) v = points[(size_t)b * CIN * NPTS + (size_t)(c - 3) * NPTS + n];
  else             v = 0.f;
  A0[t] = (_Float16)v;
}

// ---------------------------------------------------------------------------
// WMMA GEMM: Out[M,NB] = A[M,KA](f16) x Bm[KA,NB](f16), f32 accumulate.
// One 16-row M-tile per wave, all NB columns as NT=NB/16 accumulators.
// Fused per-channel sum / sum-of-squares for training-mode BatchNorm.
// Fragment layouts per CDNA5 ISA 7.12.2 (16-bit A 16x32, B lane<->K).
// ---------------------------------------------------------------------------
template <int KA, int NB>
__global__ __launch_bounds__(256) void gemm_bn_kernel(const _Float16* __restrict__ A,
                                                      const _Float16* __restrict__ Bm,
                                                      float* __restrict__ Out,
                                                      float* __restrict__ gsum,
                                                      float* __restrict__ gsq) {
  constexpr int NT = NB / 16;
  __shared__ float s_sum[128], s_sq[128];
  int tid = threadIdx.x;
  if (tid < 128) { s_sum[tid] = 0.f; s_sq[tid] = 0.f; }
  __syncthreads();

  int warp = tid >> 5, lane = tid & 31;
  int row16 = lane & 15;        // M within tile (A) / N within ntile (D)
  int hsel = lane >> 4;         // half-select for K (A) / M (D)
  const int mtiles = MROWS / 16;

  for (int mt = blockIdx.x * 8 + warp; mt < mtiles; mt += gridDim.x * 8) {
    const _Float16* Ab = A + (size_t)mt * 16 * KA;
    __builtin_prefetch(Ab + (size_t)16 * KA, 0, 1);   // global_prefetch next tile

    v8f acc[NT];
#pragma unroll
    for (int ntl = 0; ntl < NT; ++ntl) {
#pragma unroll
      for (int e = 0; e < 8; ++e) acc[ntl][e] = 0.f;
    }

#pragma unroll
    for (int kb = 0; kb < KA; kb += 32) {
      // A frag: lane m holds K=[kb+8h .. +7] and K=[kb+16+8h .. +7]
      const _Float16* ap = Ab + (size_t)row16 * KA + kb + hsel * 8;
      v8h alo = *(const v8h*)ap;
      v8h ahi = *(const v8h*)(ap + 16);
      v16h af = __builtin_shufflevector(alo, ahi, 0, 1, 2, 3, 4, 5, 6, 7,
                                        8, 9, 10, 11, 12, 13, 14, 15);
#pragma unroll
      for (int ntl = 0; ntl < NT; ++ntl) {
        // B frag: lane n = ntl*16+row16 holds K=[kb+8h .. +7] and K=[kb+16+8h .. +7]
        const _Float16* bp = Bm + (size_t)(kb + hsel * 8) * NB + ntl * 16 + row16;
        v16h bf;
#pragma unroll
        for (int s = 0; s < 8; ++s) {
          bf[s] = bp[(size_t)s * NB];
          bf[8 + s] = bp[(size_t)(16 + s) * NB];
        }
        acc[ntl] = __builtin_amdgcn_wmma_f32_16x16x32_f16(
            false, af, false, bf, (short)0, acc[ntl], false, false);
      }
    }

    // D layout: lane n = ntl*16 + row16; VGPR r -> M = 8*hsel + r
    int mo = hsel * 8;
#pragma unroll
    for (int ntl = 0; ntl < NT; ++ntl) {
      int nn = ntl * 16 + row16;
      float ls = 0.f, lq = 0.f;
      float* orow = Out + (size_t)(mt * 16 + mo) * NB + nn;
#pragma unroll
      for (int r = 0; r < 8; ++r) {
        float v = acc[ntl][r];
        orow[(size_t)r * NB] = v;
        ls += v;
        lq += v * v;
      }
      atomicAdd(&s_sum[nn], ls);   // ds_add_f32
      atomicAdd(&s_sq[nn], lq);
    }
  }
  __syncthreads();
  if (tid < NB) {
    atomicAdd(&gsum[tid], s_sum[tid]);
    atomicAdd(&gsq[tid], s_sq[tid]);
  }
}

__global__ void bn_finalize(const float* __restrict__ gsum, const float* __restrict__ gsq,
                            const float* __restrict__ gamma, const float* __restrict__ beta,
                            float* __restrict__ scale, float* __restrict__ shift,
                            int NB, float invM) {
  int c = threadIdx.x;
  if (c < NB) {
    float mean = gsum[c] * invM;
    float var = gsq[c] * invM - mean * mean;
    float sc = gamma[c] * rsqrtf(var + 1e-5f);
    scale[c] = sc;
    shift[c] = beta[c] - mean * sc;
  }
}

// BN + ReLU + cast to f16 for the next layer's A matrix (NB = 64 here).
__global__ void bn_relu_apply(const float* __restrict__ Out, const float* __restrict__ scale,
                              const float* __restrict__ shift, _Float16* __restrict__ Anext) {
  size_t t = (size_t)blockIdx.x * blockDim.x + threadIdx.x;
  if (t >= (size_t)MROWS * 64) return;
  int c = (int)(t & 63);
  float v = Out[t] * scale[c] + shift[c];
  Anext[t] = (_Float16)fmaxf(v, 0.f);
}

// Final: BN + ReLU + max over K, transposed store to [B,128,S].
__global__ void final_max_kernel(const float* __restrict__ Out2, const float* __restrict__ scale,
                                 const float* __restrict__ shift, float* __restrict__ out_np) {
  int t = blockIdx.x * blockDim.x + threadIdx.x;   // < B*S*128, grid exact
  int ch = t & 127;
  int bs = t >> 7;
  int b = bs / SPTS, s = bs % SPTS;
  float sc = scale[ch], sh = shift[ch];
  float m = 0.f;                                   // relu floor
  const float* p = Out2 + (size_t)bs * KNN * 128 + ch;
  for (int k = 0; k < KNN; ++k) {
    float v = fmaxf(p[(size_t)k * 128] * sc + sh, 0.f);
    m = fmaxf(m, v);
  }
  out_np[(size_t)b * 128 * SPTS + (size_t)ch * SPTS + s] = m;
}

// ---------------------------------------------------------------------------
extern "C" void kernel_launch(void* const* d_in, const int* in_sizes, int n_in,
                              void* d_out, int out_size, void* d_ws, size_t ws_size,
                              hipStream_t stream) {
  (void)in_sizes; (void)n_in; (void)out_size; (void)ws_size;
  const float* xyz    = (const float*)d_in[0];
  const float* points = (const float*)d_in[1];
  const float* W0 = (const float*)d_in[2];
  const float* g0 = (const float*)d_in[3];
  const float* b0 = (const float*)d_in[4];
  const float* W1 = (const float*)d_in[5];
  const float* g1 = (const float*)d_in[6];
  const float* b1 = (const float*)d_in[7];
  const float* W2 = (const float*)d_in[8];
  const float* g2 = (const float*)d_in[9];
  const float* b2 = (const float*)d_in[10];

  float* out = (float*)d_out;
  float* out_np = out + (size_t)BATCH * 3 * SPTS;

  char* w = (char*)d_ws;
  auto carve = [&](size_t bytes) -> void* {
    void* p = (void*)w;
    w += (bytes + 255) & ~(size_t)255;
    return p;
  };
  int*      fps_idx = (int*)carve((size_t)BATCH * SPTS * 4);
  float*    new_xyz = (float*)carve((size_t)BATCH * SPTS * 3 * 4);
  int*      knn_idx = (int*)carve((size_t)MROWS * 4);
  _Float16* Wt0     = (_Float16*)carve((size_t)96 * 64 * 2);
  _Float16* Wt1     = (_Float16*)carve((size_t)64 * 64 * 2);
  _Float16* Wt2     = (_Float16*)carve((size_t)64 * 128 * 2);
  float*    stats   = (float*)carve((size_t)6 * 128 * 4);   // [sum,sq] x 3 layers
  float*    scsh    = (float*)carve((size_t)6 * 128 * 4);   // [scale,shift] x 3 layers
  _Float16* A0      = (_Float16*)carve((size_t)MROWS * 96 * 2);
  _Float16* A1      = (_Float16*)carve((size_t)MROWS * 64 * 2);
  _Float16* A2      = A0;                                   // A0 dead after gemm0
  float*    outbuf  = (float*)carve((size_t)MROWS * 128 * 4);

  const float invM = 1.0f / (float)MROWS;

  convert_weights<<<(96 * 64 + 64 * 64 + 64 * 128 + 255) / 256, 256, 0, stream>>>(
      W0, W1, W2, Wt0, Wt1, Wt2);
  zero_f32<<<3, 256, 0, stream>>>(stats, 6 * 128);

  fps_kernel<<<BATCH, 256, 0, stream>>>(xyz, fps_idx, new_xyz, out);
  knn_kernel<<<(BATCH * SPTS) / 8, 256, 0, stream>>>(xyz, new_xyz, knn_idx);
  gather_kernel<<<(int)(((size_t)MROWS * 96 + 255) / 256), 256, 0, stream>>>(
      xyz, points, new_xyz, knn_idx, A0);

  // Layer 0: [M,96] x [96,64]
  gemm_bn_kernel<96, 64><<<4096, 256, 0, stream>>>(A0, Wt0, outbuf, stats + 0, stats + 128);
  bn_finalize<<<1, 128, 0, stream>>>(stats + 0, stats + 128, g0, b0, scsh + 0, scsh + 128, 64, invM);
  bn_relu_apply<<<(int)(((size_t)MROWS * 64 + 255) / 256), 256, 0, stream>>>(
      outbuf, scsh + 0, scsh + 128, A1);

  // Layer 1: [M,64] x [64,64]
  gemm_bn_kernel<64, 64><<<4096, 256, 0, stream>>>(A1, Wt1, outbuf, stats + 256, stats + 384);
  bn_finalize<<<1, 128, 0, stream>>>(stats + 256, stats + 384, g1, b1, scsh + 256, scsh + 384, 64, invM);
  bn_relu_apply<<<(int)(((size_t)MROWS * 64 + 255) / 256), 256, 0, stream>>>(
      outbuf, scsh + 256, scsh + 384, A2);

  // Layer 2: [M,64] x [64,128]
  gemm_bn_kernel<64, 128><<<4096, 256, 0, stream>>>(A2, Wt2, outbuf, stats + 512, stats + 640);
  bn_finalize<<<1, 128, 0, stream>>>(stats + 512, stats + 640, g2, b2, scsh + 512, scsh + 640, 128, invM);
  final_max_kernel<<<(BATCH * SPTS * 128) / 256, 256, 0, stream>>>(
      outbuf, scsh + 512, scsh + 640, out_np);
}
